// GenDecAttnBlock_52441550684836
// MI455X (gfx1250) — matrix operand order repaired and checked by reading, not compiled, not run on hardware
//
#include <hip/hip_runtime.h>

// ---------- CDNA5 WMMA types & helpers ----------
typedef __attribute__((ext_vector_type(16))) __bf16 v16bf;
typedef __attribute__((ext_vector_type(8)))  __bf16 bf16x8;
typedef __attribute__((ext_vector_type(8)))  float  v8f;
typedef unsigned short u16b;

#define DEVI static __device__ __forceinline__

DEVI u16b f2bfbits(float f) {
  unsigned u = __builtin_bit_cast(unsigned, f);
  return (u16b)((u + 0x7FFFu + ((u >> 16) & 1u)) >> 16);  // RNE truncate to bf16
}
DEVI __bf16 bfraw(u16b s) { return __builtin_bit_cast(__bf16, s); }
DEVI v8f vzero8() { v8f z;
#pragma unroll
  for (int i = 0; i < 8; ++i) z[i] = 0.f;
  return z; }
DEVI bf16x8 bzero8() { bf16x8 z;
#pragma unroll
  for (int i = 0; i < 8; ++i) z[i] = bfraw((u16b)0);
  return z; }
// 16-byte vector load of 8 bf16 (emits global/ds b128 loads)
DEVI bf16x8 ld8(const u16b* p) { return *(const bf16x8*)p; }
DEVI v16bf mk16(bf16x8 lo, bf16x8 hi) {
  v16bf r;
#pragma unroll
  for (int i = 0; i < 8; ++i) { r[i] = lo[i]; r[i + 8] = hi[i]; }
  return r;
}

// Fragment fetch helpers, contiguous K order (k = 16*hf + s) shared by A and B.
// A (16x32): lane lr = row M; loads K in [16*hf, 16*hf+16)
DEVI v16bf loadA(const u16b* row, int c0, int hf) {
  return mk16(ld8(row + c0 + 16 * hf), ld8(row + c0 + 16 * hf + 8));
}
// B (32x16): lane lr = col N; loads K in [16*hf, 16*hf+16)
DEVI v16bf loadB(const u16b* col, int c0, int hf) {
  return mk16(ld8(col + c0 + 16 * hf), ld8(col + c0 + 16 * hf + 8));
}

#define WMMA_BF16(A, B, C) \
  __builtin_amdgcn_wmma_f32_16x16x32_bf16(false, (A), false, (B), (short)0, (C), false, false)

// Problem constants: B=4, C=256, H=W=64, HW=4096, d=128, prev_c=512, concat=768

// ============================================================================
// K0a: pack Wq/Wk/Wv/Wp to bf16 (same row-major M x K layout)
// ============================================================================
__global__ void pack_qkvp_kernel(const float* __restrict__ Wq, const float* __restrict__ Wk,
                                 const float* __restrict__ Wv, const float* __restrict__ Wp,
                                 u16b* __restrict__ Wqb, u16b* __restrict__ Wkb,
                                 u16b* __restrict__ Wvb, u16b* __restrict__ Wpb) {
  int idx = blockIdx.x * 256 + threadIdx.x;  // 4 * 32768
  int sel = idx >> 15, i = idx & 32767;
  if (sel == 0)      Wqb[i] = f2bfbits(Wq[i]);
  else if (sel == 1) Wkb[i] = f2bfbits(Wk[i]);
  else if (sel == 2) Wvb[i] = f2bfbits(Wv[i]);
  else               Wpb[i] = f2bfbits(Wp[i]);
}

// ============================================================================
// K0b: repack Wc (co,ci,ky,kx) fp32 -> (tap, co, ci) bf16 so conv A-fragments
//      are contiguous over ci
// ============================================================================
__global__ void pack_wc_kernel(const float* __restrict__ Wc, u16b* __restrict__ Wcb) {
  int idx = blockIdx.x * 256 + threadIdx.x;  // 256*768*9 = 1769472
  int co = idx / 6912;
  int r = idx - co * 6912;
  int ci = r / 9;
  int tap = r - ci * 9;
  Wcb[((size_t)tap * 256 + co) * 768 + ci] = f2bfbits(Wc[idx]);
}

// ============================================================================
// K0c: transpose activations (B,C,HW) fp32 -> pixel-major (B,HW,C) bf16 so the
//      projection B-fragments are contiguous b128 loads (no LDS staging)
// ============================================================================
__global__ void pack_x_kernel(const float* __restrict__ src, const float* __restrict__ tgt,
                              u16b* __restrict__ srcT, u16b* __restrict__ tgtT) {
  size_t idx = (size_t)blockIdx.x * 256 + threadIdx.x;  // 2 * 4*256*4096
  int sel = (int)(idx >> 22);
  size_t i = idx & ((1u << 22) - 1);
  int b = (int)(i >> 20);
  int r = (int)(i & 1048575);
  int ch = r >> 12;
  int p = r & 4095;
  const float* X = sel ? tgt : src;
  u16b* O = sel ? tgtT : srcT;
  O[((size_t)b * 4096 + p) * 256 + ch] = f2bfbits(X[i]);
}

// ============================================================================
// K1: q/k/v projections (WMMA GEMM, M=128, K=256, N=4096 per batch)
//     pure b128-load + WMMA (operands pre-packed bf16, K-contiguous)
//     q,k stored (HW,d) bf16; v stored (d,HW) bf16
// ============================================================================
__global__ void qkv_proj_kernel(const u16b* __restrict__ srcT, const u16b* __restrict__ tgtT,
                                const u16b* __restrict__ Wqb, const u16b* __restrict__ Wkb,
                                const u16b* __restrict__ Wvb,
                                const float* __restrict__ bq, const float* __restrict__ bk,
                                const float* __restrict__ bv,
                                u16b* __restrict__ qT, u16b* __restrict__ kT,
                                u16b* __restrict__ vT) {
  const int tid = threadIdx.x;
  const int lane = tid & 31, wave = tid >> 5;
  const int hf = lane >> 4, lr = lane & 15;
  const int n0 = blockIdx.x * 16;
  const int b = blockIdx.y;
  const int mode = blockIdx.z;  // 0=q(src) 1=k(tgt) 2=v(tgt)
  const u16b* X = (mode == 0) ? srcT : tgtT;
  const u16b* W = (mode == 0) ? Wqb : (mode == 1) ? Wkb : Wvb;
  const float* bias = (mode == 0) ? bq : (mode == 1) ? bk : bv;
  const int m0 = wave * 16;  // 8 waves cover M=128
  const u16b* wrow = W + (m0 + lr) * 256;                     // A: K-contiguous
  const u16b* xrow = X + ((size_t)b * 4096 + n0 + lr) * 256;  // B: K-contiguous

  v8f acc = vzero8();
#pragma unroll
  for (int k0 = 0; k0 < 256; k0 += 32) {
    acc = WMMA_BF16(loadA(wrow, k0, hf), loadB(xrow, k0, hf), acc);
  }
#pragma unroll
  for (int r = 0; r < 8; ++r) {
    int row = m0 + r + 8 * hf;
    int col = n0 + lr;
    u16b v = f2bfbits(acc[r] + bias[row]);
    if (mode == 0)      qT[((size_t)b * 4096 + col) * 128 + row] = v;
    else if (mode == 1) kT[((size_t)b * 4096 + col) * 128 + row] = v;
    else                vT[((size_t)b * 128 + row) * 4096 + col] = v;
  }
}

// ============================================================================
// K2: per-column softmax stats over key axis i:  m[j]=max_i s[i,j], l[j]=sum exp
//     score tiles recomputed via WMMA (never hit HBM)
// ============================================================================
__global__ void colstats_kernel(const u16b* __restrict__ kT, const u16b* __restrict__ qT,
                                float* __restrict__ mcol, float* __restrict__ lcol) {
  __shared__ float ms[256], ls[256];
  const int tid = threadIdx.x, lane = tid & 31, wave = tid >> 5;
  const int hf = lane >> 4, lr = lane & 15;
  const int j0 = blockIdx.x * 16;
  const int b = blockIdx.y;
  const u16b* qcol = qT + ((size_t)b * 4096 + j0 + lr) * 128;  // B operand: K-major
  float mr = -3.0e38f, lrun = 0.f;

  for (int i0 = wave * 16; i0 < 4096; i0 += 128) {
    const u16b* krow = kT + ((size_t)b * 4096 + i0 + lr) * 128;  // A operand
    v8f s = vzero8();
#pragma unroll
    for (int c0 = 0; c0 < 128; c0 += 32) {
      s = WMMA_BF16(loadA(krow, c0, hf), loadB(qcol, c0, hf), s);
    }
#pragma unroll
    for (int r = 0; r < 8; ++r) {  // online max/sum-exp for this lane's column
      float sv = s[r];
      float mn = fmaxf(mr, sv);
      lrun = lrun * __expf(mr - mn) + __expf(sv - mn);
      mr = mn;
    }
  }
  ms[tid] = mr; ls[tid] = lrun;
  __syncthreads();
  if (tid < 16) {  // combine 8 waves x 2 half-lanes per column
    float M = -3.0e38f, L = 0.f;
    for (int t = 0; t < 16; ++t) {
      float m2 = ms[tid + 16 * t], l2 = ls[tid + 16 * t];
      float mn = fmaxf(M, m2);
      L = L * __expf(M - mn) + l2 * __expf(m2 - mn);
      M = mn;
    }
    mcol[(size_t)b * 4096 + j0 + tid] = M;
    lcol[(size_t)b * 4096 + j0 + tid] = L;
  }
}

// ============================================================================
// K3: out[i,c] = sum_j exp(s[i,j]-m[j])/l[j] * v[j,c]  (two-pass flash style)
//     P tile goes through LDS for the C-layout -> A-layout cross-lane shuffle
// ============================================================================
__global__ void attn_out_kernel(const u16b* __restrict__ kT, const u16b* __restrict__ qT,
                                const u16b* __restrict__ vT,
                                const float* __restrict__ mcol, const float* __restrict__ lcol,
                                u16b* __restrict__ oab) {
  __shared__ __align__(16) u16b pl[8][16][32];  // per-wave P tile (16 i x 32 j)
  const int tid = threadIdx.x, lane = tid & 31, wave = tid >> 5;
  const int hf = lane >> 4, lr = lane & 15;
  const int b = blockIdx.y;
  const int i0 = blockIdx.x * 128 + wave * 16;
  const u16b* krow = kT + ((size_t)b * 4096 + i0 + lr) * 128;

  v8f acc[8];
#pragma unroll
  for (int t = 0; t < 8; ++t) acc[t] = vzero8();

  for (int j0 = 0; j0 < 4096; j0 += 32) {
    const u16b* qcol0 = qT + ((size_t)b * 4096 + j0 + lr) * 128;
    const u16b* qcol1 = qcol0 + 16 * 128;
    v8f s0 = vzero8(), s1 = vzero8();
#pragma unroll
    for (int c0 = 0; c0 < 128; c0 += 32) {
      v16bf a = loadA(krow, c0, hf);
      s0 = WMMA_BF16(a, loadB(qcol0, c0, hf), s0);
      s1 = WMMA_BF16(a, loadB(qcol1, c0, hf), s1);
    }
    float m0c = mcol[(size_t)b * 4096 + j0 + lr];
    float rl0 = 1.0f / lcol[(size_t)b * 4096 + j0 + lr];
    float m1c = mcol[(size_t)b * 4096 + j0 + 16 + lr];
    float rl1 = 1.0f / lcol[(size_t)b * 4096 + j0 + 16 + lr];
#pragma unroll
    for (int r = 0; r < 8; ++r) {
      int row = r + 8 * hf;
      pl[wave][row][lr] = f2bfbits(__expf(s0[r] - m0c) * rl0);
      pl[wave][row][16 + lr] = f2bfbits(__expf(s1[r] - m1c) * rl1);
    }
    __syncthreads();
    v16bf pa = loadA(&pl[wave][lr][0], 0, hf);  // ds_load_b128 x2
#pragma unroll
    for (int ct = 0; ct < 8; ++ct) {
      const u16b* vrow = vT + ((size_t)b * 128 + ct * 16 + lr) * 4096;  // B: K=j contiguous
      acc[ct] = WMMA_BF16(pa, loadB(vrow, j0, hf), acc[ct]);
    }
    __syncthreads();
  }
#pragma unroll
  for (int ct = 0; ct < 8; ++ct)
#pragma unroll
    for (int r = 0; r < 8; ++r) {
      int i = i0 + r + 8 * hf;
      oab[((size_t)b * 4096 + i) * 128 + ct * 16 + lr] = f2bfbits(acc[ct][r]);
    }
}

// ============================================================================
// K4: out-projection Wp (256x128) + residual*gamma, write bf16 into the
//     pixel-major concat buffer channels [0,256)
// ============================================================================
__global__ void proj_residual_kernel(const u16b* __restrict__ Wpb, const float* __restrict__ bp,
                                     const u16b* __restrict__ oab, const float* __restrict__ tgt,
                                     const float* __restrict__ gamma, u16b* __restrict__ ccT) {
  const int tid = threadIdx.x, lane = tid & 31, wave = tid >> 5;
  const int hf = lane >> 4, lr = lane & 15;
  const int n0 = blockIdx.x * 16;
  const int b = blockIdx.y;
  const int m0 = blockIdx.z * 128 + wave * 16;
  const u16b* wrow = Wpb + (m0 + lr) * 128;
  const u16b* ocol = oab + ((size_t)b * 4096 + n0 + lr) * 128;

  v8f acc = vzero8();
#pragma unroll
  for (int c0 = 0; c0 < 128; c0 += 32) {
    acc = WMMA_BF16(loadA(wrow, c0, hf), loadB(ocol, c0, hf), acc);
  }
  float g = gamma[0];
#pragma unroll
  for (int r = 0; r < 8; ++r) {
    int row = m0 + r + 8 * hf;
    int col = n0 + lr;
    float val = acc[r] + bp[row];
    float x = tgt[((size_t)b * 256 + row) * 4096 + col] + g * val;
    ccT[((size_t)b * 4096 + col) * 768 + row] = f2bfbits(x);
  }
}

// ============================================================================
// K5: nearest 2x upsample of prev into concat channels [256,768), pixel-major
// ============================================================================
__global__ void upsample_concat_kernel(const float* __restrict__ prev, u16b* __restrict__ ccT) {
  size_t idx = (size_t)blockIdx.x * blockDim.x + threadIdx.x;  // 4*512*4096
  if (idx >= (size_t)4 * 512 * 4096) return;
  int x = (int)(idx & 63);
  int y = (int)((idx >> 6) & 63);
  int pc = (int)((idx >> 12) & 511);
  int b = (int)(idx >> 21);
  float v = prev[(((size_t)b * 512 + pc) * 32 + (y >> 1)) * 32 + (x >> 1)];
  ccT[((size_t)b * 4096 + y * 64 + x) * 768 + 256 + pc] = f2bfbits(v);
}

// ============================================================================
// K6: 3x3 conv (stride 1, pad 1) as implicit GEMM: M=256, K=768*9, N=4096/batch
//     A from repacked (tap,co,ci) bf16 weights; B from pixel-major concat
// ============================================================================
__global__ void conv3x3_kernel(const u16b* __restrict__ ccT, const u16b* __restrict__ Wcb,
                               const float* __restrict__ bc, float* __restrict__ cv) {
  const int tid = threadIdx.x, lane = tid & 31, wave = tid >> 5;
  const int hf = lane >> 4, lr = lane & 15;
  const int n0 = blockIdx.x * 16;  // 16 pixels on one output row
  const int m0 = blockIdx.y * 128 + wave * 16;
  const int b = blockIdx.z;
  const int y = n0 >> 6;
  const int x0 = n0 & 63;

  v8f acc = vzero8();
  for (int tap = 0; tap < 9; ++tap) {
    int ky = tap / 3, kx = tap % 3;
    int yy = y + ky - 1;
    int xx = x0 + lr + kx - 1;
    bool valid = ((unsigned)yy < 64u) && ((unsigned)xx < 64u);
    const u16b* wrow = Wcb + ((size_t)tap * 256 + (m0 + lr)) * 768;
    const u16b* bcol = ccT + ((size_t)b * 4096 + yy * 64 + xx) * 768;
    for (int ci0 = 0; ci0 < 768; ci0 += 32) {
      if (ci0 + 32 < 768) __builtin_prefetch(bcol + ci0 + 32, 0, 1);
      v16bf a = loadA(wrow, ci0, hf);
      bf16x8 blo = bzero8(), bhi = bzero8();
      if (valid) {
        blo = ld8(bcol + ci0 + 16 * hf);
        bhi = ld8(bcol + ci0 + 16 * hf + 8);
      }
      acc = WMMA_BF16(a, mk16(blo, bhi), acc);  // uniform control flow here
    }
  }
#pragma unroll
  for (int r = 0; r < 8; ++r) {
    int co = m0 + r + 8 * hf;
    cv[((size_t)b * 256 + co) * 4096 + n0 + lr] = acc[r] + bc[co];
  }
}

// ============================================================================
// K7: InstanceNorm (biased var over HW, eps=1e-5) + ReLU, one block per (b,c)
// ============================================================================
__global__ void instnorm_relu_kernel(const float* __restrict__ cv, float* __restrict__ out) {
  __shared__ float s1[256], s2[256];
  const int tid = threadIdx.x;
  const size_t base = (size_t)blockIdx.x * 4096;  // blockIdx.x = b*256+c
  float sm = 0.f, sq = 0.f;
  for (int t = tid; t < 4096; t += 256) {
    float v = cv[base + t];
    sm += v; sq += v * v;
  }
  s1[tid] = sm; s2[tid] = sq;
  __syncthreads();
  for (int off = 128; off > 0; off >>= 1) {
    if (tid < off) { s1[tid] += s1[tid + off]; s2[tid] += s2[tid + off]; }
    __syncthreads();
  }
  float mean = s1[0] * (1.0f / 4096.0f);
  float var = s2[0] * (1.0f / 4096.0f) - mean * mean;
  float inv = rsqrtf(var + 1e-5f);
  for (int t = tid; t < 4096; t += 256) {
    float v = (cv[base + t] - mean) * inv;
    out[base + t] = fmaxf(v, 0.f);
  }
}

// ============================================================================
extern "C" void kernel_launch(void* const* d_in, const int* in_sizes, int n_in,
                              void* d_out, int out_size, void* d_ws, size_t ws_size,
                              hipStream_t stream) {
  (void)in_sizes; (void)n_in; (void)out_size; (void)ws_size;
  const float* src = (const float*)d_in[0];
  const float* tgt = (const float*)d_in[1];
  const float* prev = (const float*)d_in[2];
  const float* Wq = (const float*)d_in[3];  const float* bq = (const float*)d_in[4];
  const float* Wk = (const float*)d_in[5];  const float* bk = (const float*)d_in[6];
  const float* Wv = (const float*)d_in[7];  const float* bv = (const float*)d_in[8];
  const float* Wp = (const float*)d_in[9];  const float* bp = (const float*)d_in[10];
  const float* gamma = (const float*)d_in[11];
  const float* Wc = (const float*)d_in[12]; const float* bc = (const float*)d_in[13];
  float* out = (float*)d_out;

  // workspace layout (bytes), ~62.7 MB total
  char* ws = (char*)d_ws;
  u16b*  qT   = (u16b*)(ws + 0);          // (B,4096,128) bf16   4 MB  (K-major for B frags)
  u16b*  kT   = (u16b*)(ws + 4194304);    // (B,4096,128) bf16   4 MB  (rows for A frags)
  u16b*  vT   = (u16b*)(ws + 8388608);    // (B,128,4096) bf16   4 MB  (K=j contiguous)
  float* mcol = (float*)(ws + 12582912);  // (B,4096) f32        64 KB
  float* lcol = (float*)(ws + 12648448);  // (B,4096) f32        64 KB
  u16b*  oab  = (u16b*)(ws + 12713984);   // (B,4096,128) bf16   4 MB
  u16b*  ccT  = (u16b*)(ws + 16908288);   // (B,4096,768) bf16   24 MB (pixel-major)
  float* cv   = (float*)(ws + 42074112);  // (B,256,4096) f32    16 MB
  u16b*  Wqb  = (u16b*)(ws + 58851328);   // 128x256 bf16        64 KB
  u16b*  Wkb  = (u16b*)(ws + 58916864);
  u16b*  Wvb  = (u16b*)(ws + 58982400);
  u16b*  Wpb  = (u16b*)(ws + 59047936);   // 256x128 bf16        64 KB
  u16b*  Wcb  = (u16b*)(ws + 59113472);   // (9,256,768) bf16    3.4 MB -> end 62652416
  // transient: transposed bf16 activations live in the (not-yet-written) ccT
  // region; they are fully consumed by K1 before K4/K5 overwrite ccT.
  u16b*  srcT = (u16b*)(ws + 16908288);   // (B,4096,256) bf16   8 MB
  u16b*  tgtT = (u16b*)(ws + 25296896);   // (B,4096,256) bf16   8 MB

  pack_qkvp_kernel<<<512, 256, 0, stream>>>(Wq, Wk, Wv, Wp, Wqb, Wkb, Wvb, Wpb);
  pack_wc_kernel<<<6912, 256, 0, stream>>>(Wc, Wcb);
  pack_x_kernel<<<32768, 256, 0, stream>>>(src, tgt, srcT, tgtT);
  qkv_proj_kernel<<<dim3(256, 4, 3), 256, 0, stream>>>(srcT, tgtT, Wqb, Wkb, Wvb, bq, bk, bv,
                                                       qT, kT, vT);
  colstats_kernel<<<dim3(256, 4), 256, 0, stream>>>(kT, qT, mcol, lcol);
  attn_out_kernel<<<dim3(32, 4), 256, 0, stream>>>(kT, qT, vT, mcol, lcol, oab);
  proj_residual_kernel<<<dim3(256, 4, 2), 256, 0, stream>>>(Wpb, bp, oab, tgt, gamma, ccT);
  upsample_concat_kernel<<<32768, 256, 0, stream>>>(prev, ccT);
  conv3x3_kernel<<<dim3(256, 2, 4), 256, 0, stream>>>(ccT, Wcb, bc, cv);
  instnorm_relu_kernel<<<1024, 256, 0, stream>>>(cv, out);
}
